// CollConv_74019466379556
// MI455X (gfx1250) — compile-verified, matrix operand order checked by reading, not run on hardware
//
#include <hip/hip_runtime.h>
#include <hip/hip_bf16.h>
#include <math.h>

#define N_NODES 50000
#define N_EDGES 800000
#define D_IN 96
#define H1DIM 32
#define H2DIM 64
#define D_OUT 128
#define NTILES (N_NODES / 16)   // 3125 exact
#define BN_EPS 1e-5f
#define LRELU 0.01f

typedef __attribute__((ext_vector_type(16))) _Float16 v16h;
typedef __attribute__((ext_vector_type(8)))  float    v8f;

// Hardware transcendental sigmoid: v_exp_f32 + v_rcp_f32 (TRANS pipe,
// co-executes with WMMA on CDNA5).
__device__ __forceinline__ float fast_sigmoid(float x) {
    float e = __builtin_amdgcn_exp2f(x * -1.44269504088896341f);
    return __builtin_amdgcn_rcpf(1.0f + e);
}

// ---------------------------------------------------------------------------
// Kernel 1: agg[n][d] = x[n][d]  (GIN eps=0: h = x + sum_neighbors).
// Also zero the 256-float BN stats buffer.
// ---------------------------------------------------------------------------
__global__ void init_kernel(const float* __restrict__ x, float* __restrict__ agg,
                            float* __restrict__ gstats) {
    size_t i = (size_t)blockIdx.x * blockDim.x + threadIdx.x;
    if (i < (size_t)N_NODES * D_IN) agg[i] = x[i];
    if (blockIdx.x == 0 && threadIdx.x < 256) gstats[threadIdx.x] = 0.0f;
}

// ---------------------------------------------------------------------------
// Kernel 2: edge scatter-add. One wave32 per edge; 96 features in 3 strides.
// Gather table (19.2 MB) and accumulator both L2-resident on the 192MB L2.
// ---------------------------------------------------------------------------
__global__ void edge_scatter_kernel(const float* __restrict__ x,
                                    const long long* __restrict__ ei,
                                    float* __restrict__ agg) {
    int wid  = (int)(((size_t)blockIdx.x * blockDim.x + threadIdx.x) >> 5);
    int lane = threadIdx.x & 31;
    if (wid >= N_EDGES) return;
    long long s = ei[wid];
    long long d = ei[N_EDGES + wid];
    const float* xs = x   + (size_t)s * D_IN;
    float*       ad = agg + (size_t)d * D_IN;
    #pragma unroll
    for (int f = lane; f < D_IN; f += 32)
        atomicAdd(&ad[f], xs[f]);
}

// ---------------------------------------------------------------------------
// WMMA fragment layout helpers (wave32, v_wmma_*_16x16x32 operand layout):
//  - A (16x32 f16): lane<16 holds row M=lane, K in {0..7, 16..23};
//    lane>=16 holds row M=lane-16, K in {8..15, 24..31}.
//    half index i -> k = (i<8) ? kb+i : 16+kb+(i-8), kb = (lane&16)?8:0.
//  - B mirrors A with N = lane&15.
//  - C/D f32: element r -> M = r + 8*(lane>=16), N = lane&15.
// Swizzled storage: frag f stored as [f][lane][16 halfs] (32B/lane, so a
// fragment load is two ds_load_b128 per lane).
// ---------------------------------------------------------------------------
__device__ __forceinline__ void stage_swizzled(const float* __restrict__ W,
                                               _Float16* __restrict__ dst,
                                               int K, int N, int tid) {
    const int total = K * N;   // total halfs = (K/32)*(N/16)*32*16
    const int ntn = N >> 4;
    for (int idx = tid; idx < total; idx += 256) {
        int i    = idx & 15;
        int lane = (idx >> 4) & 31;
        int f    = idx >> 9;
        int nt   = f % ntn;
        int kt   = f / ntn;
        int kb   = (lane & 16) ? 8 : 0;
        int k    = kt * 32 + ((i < 8) ? (kb + i) : (8 + kb + i));
        int col  = nt * 16 + (lane & 15);
        dst[idx] = (_Float16)W[k * N + col];
    }
}

__device__ __forceinline__ v16h frag_ld(const _Float16* base, int f, int lane) {
    return *(const v16h*)(base + (f << 9) + (lane << 4));
}

// Producer side: place activation value h[m][kk] (kk in [0,32) within a
// k-tile) where the A-fragment consumer expects it.
__device__ __forceinline__ void act_store(_Float16* buf, int m, int kk, _Float16 v) {
    int laneT = m + (((kk >> 3) & 1) ? 16 : 0);
    int iT    = (kk & 7) + ((kk >> 4) ? 8 : 0);
    buf[laneT * 16 + iT] = v;
}

// ---------------------------------------------------------------------------
// Kernel 3: fused MLP via v_wmma_f32_16x16x32_f16.
// 256 threads = 8 waves; each wave owns one 16-node tile.
// ---------------------------------------------------------------------------
__global__ void __launch_bounds__(256)
mlp_wmma_kernel(const float* __restrict__ agg,
                const float* __restrict__ W1, const float* __restrict__ b1,
                const float* __restrict__ W2, const float* __restrict__ b2,
                const float* __restrict__ W3, const float* __restrict__ b3,
                float* __restrict__ out,
                float* __restrict__ gsum, float* __restrict__ gsq) {
    __shared__ __align__(32) _Float16 sW1[D_IN * H1DIM];    // swizzled frags
    __shared__ __align__(32) _Float16 sW2[H1DIM * H2DIM];
    __shared__ __align__(32) _Float16 sW3[H2DIM * D_OUT];
    __shared__ __align__(32) _Float16 sH1[8][16 * H1DIM];   // A-frag layout
    __shared__ __align__(32) _Float16 sH2[8][16 * H2DIM];   // A-frag layout
    __shared__ float sSum[D_OUT];
    __shared__ float sSq[D_OUT];

    const int tid = threadIdx.x;
    stage_swizzled(W1, sW1, D_IN,  H1DIM, tid);   // frags: kt(3) x nt(2)
    stage_swizzled(W2, sW2, H1DIM, H2DIM, tid);   // frags: nt(4)
    stage_swizzled(W3, sW3, H2DIM, D_OUT, tid);   // frags: kt(2) x nt(8)
    if (tid < D_OUT) { sSum[tid] = 0.0f; sSq[tid] = 0.0f; }
    __syncthreads();

    const int wave = tid >> 5;
    const int lane = tid & 31;
    const int tile = blockIdx.x * 8 + wave;

    if (tile < NTILES) {                        // wave-uniform: EXEC all-ones
        const int row  = lane & 15;
        const int kb   = (lane & 16) ? 8 : 0;
        const int mhi  = (lane & 16) ? 8 : 0;
        const int node0 = tile * 16;
        const float* h0 = agg + (size_t)(node0 + row) * D_IN;
        _Float16* Hl1 = sH1[wave];
        _Float16* Hl2 = sH2[wave];

        // ---- layer-1 A fragments straight from global: per k-tile the
        // lane's halfs are two contiguous 8-float runs -> 4x b128 loads ----
        v16h a1[3];
        #pragma unroll
        for (int kt = 0; kt < 3; ++kt) {
            const float4* p0 = (const float4*)(h0 + kt * 32 + kb);
            const float4* p1 = (const float4*)(h0 + kt * 32 + 16 + kb);
            float4 q0 = p0[0], q1 = p0[1], q2 = p1[0], q3 = p1[1];
            v16h a;
            a[0]  = (_Float16)q0.x; a[1]  = (_Float16)q0.y;
            a[2]  = (_Float16)q0.z; a[3]  = (_Float16)q0.w;
            a[4]  = (_Float16)q1.x; a[5]  = (_Float16)q1.y;
            a[6]  = (_Float16)q1.z; a[7]  = (_Float16)q1.w;
            a[8]  = (_Float16)q2.x; a[9]  = (_Float16)q2.y;
            a[10] = (_Float16)q2.z; a[11] = (_Float16)q2.w;
            a[12] = (_Float16)q3.x; a[13] = (_Float16)q3.y;
            a[14] = (_Float16)q3.z; a[15] = (_Float16)q3.w;
            a1[kt] = a;
        }

        // -------- layer 1: [16x96] @ [96x32] + b1, sigmoid --------
        #pragma unroll
        for (int nt = 0; nt < H1DIM / 16; ++nt) {
            v8f c = {};
            #pragma unroll
            for (int kt = 0; kt < 3; ++kt) {
                v16h b = frag_ld(sW1, kt * 2 + nt, lane);
                c = __builtin_amdgcn_wmma_f32_16x16x32_f16(
                        false, a1[kt], false, b, (short)0, c, false, false);
            }
            #pragma unroll
            for (int r = 0; r < 8; ++r) {
                const int m = r + mhi;
                const int n = nt * 16 + row;
                act_store(Hl1, m, n, (_Float16)fast_sigmoid(c[r] + b1[n]));
            }
        }

        // -------- layer 2: [16x32] @ [32x64] + b2, sigmoid --------
        v16h a2 = *(const v16h*)(Hl1 + lane * 16);   // single K-tile
        #pragma unroll
        for (int nt = 0; nt < H2DIM / 16; ++nt) {
            v16h b = frag_ld(sW2, nt, lane);
            v8f c = {};
            c = __builtin_amdgcn_wmma_f32_16x16x32_f16(
                    false, a2, false, b, (short)0, c, false, false);
            #pragma unroll
            for (int r = 0; r < 8; ++r) {
                const int m = r + mhi;
                const int n = nt * 16 + row;
                act_store(Hl2 + ((n >> 5) << 9), m, n & 31,
                          (_Float16)fast_sigmoid(c[r] + b2[n]));
            }
        }

        // -------- layer 3: [16x64] @ [64x128] + b3, LeakyReLU, stats --------
        v16h a3[2];
        a3[0] = *(const v16h*)(Hl2 + lane * 16);
        a3[1] = *(const v16h*)(Hl2 + 512 + lane * 16);
        #pragma unroll
        for (int nt = 0; nt < D_OUT / 16; ++nt) {
            v8f c = {};
            #pragma unroll
            for (int kt = 0; kt < 2; ++kt) {
                v16h b = frag_ld(sW3, kt * 8 + nt, lane);
                c = __builtin_amdgcn_wmma_f32_16x16x32_f16(
                        false, a3[kt], false, b, (short)0, c, false, false);
            }
            #pragma unroll
            for (int r = 0; r < 8; ++r) {
                const int m = r + mhi;
                const int n = nt * 16 + row;
                float v = c[r] + b3[n];
                v = (v >= 0.0f) ? v : (LRELU * v);
                out[(size_t)(node0 + m) * D_OUT + n] = v;
                atomicAdd(&sSum[n], v);        // ds_add_f32
                atomicAdd(&sSq[n],  v * v);
            }
        }
    }

    __syncthreads();
    if (tid < D_OUT) {
        atomicAdd(&gsum[tid], sSum[tid]);
        atomicAdd(&gsq[tid],  sSq[tid]);
    }
}

// ---------------------------------------------------------------------------
// Kernel 4: BatchNorm normalize in place (biased variance, training mode).
// ---------------------------------------------------------------------------
__global__ void bn_kernel(float* __restrict__ out,
                          const float* __restrict__ gsum,
                          const float* __restrict__ gsq,
                          const float* __restrict__ gamma,
                          const float* __restrict__ beta) {
    size_t i = (size_t)blockIdx.x * blockDim.x + threadIdx.x;
    if (i >= (size_t)N_NODES * D_OUT) return;
    const int c = (int)(i & (D_OUT - 1));
    const float invN = 1.0f / (float)N_NODES;
    const float mean = gsum[c] * invN;
    const float var  = gsq[c] * invN - mean * mean;
    const float v = out[i];
    out[i] = (v - mean) * __builtin_amdgcn_rsqf(var + BN_EPS) * gamma[c] + beta[c];
}

// ---------------------------------------------------------------------------
extern "C" void kernel_launch(void* const* d_in, const int* in_sizes, int n_in,
                              void* d_out, int out_size, void* d_ws, size_t ws_size,
                              hipStream_t stream) {
    const float*     x     = (const float*)d_in[0];
    const long long* ei    = (const long long*)d_in[1];
    const float*     W1    = (const float*)d_in[2];
    const float*     b1    = (const float*)d_in[3];
    const float*     W2    = (const float*)d_in[4];
    const float*     b2    = (const float*)d_in[5];
    const float*     W3    = (const float*)d_in[6];
    const float*     b3    = (const float*)d_in[7];
    const float*     gamma = (const float*)d_in[8];
    const float*     beta  = (const float*)d_in[9];

    float* agg    = (float*)d_ws;                                  // 50000*96 f32
    float* gstats = (float*)((char*)d_ws + (size_t)N_NODES * D_IN * sizeof(float));
    float* gsum   = gstats;          // 128
    float* gsq    = gstats + D_OUT;  // 128
    float* out    = (float*)d_out;

    {
        const int total = N_NODES * D_IN;
        init_kernel<<<(total + 255) / 256, 256, 0, stream>>>(x, agg, gstats);
    }
    {
        const int blocks = (N_EDGES + 7) / 8;   // one wave per edge
        edge_scatter_kernel<<<blocks, 256, 0, stream>>>(x, ei, agg);
    }
    {
        const int blocks = (NTILES + 7) / 8;
        mlp_wmma_kernel<<<blocks, 256, 0, stream>>>(agg, W1, b1, W2, b2, W3, b3,
                                                    out, gsum, gsq);
    }
    {
        const size_t total = (size_t)N_NODES * D_OUT;
        bn_kernel<<<(unsigned)((total + 255) / 256), 256, 0, stream>>>(
            out, gsum, gsq, gamma, beta);
    }
    // tuple output: append edge_index bytes after the node features
    hipMemcpyAsync((char*)d_out + (size_t)N_NODES * D_OUT * sizeof(float),
                   (const void*)ei, (size_t)2 * N_EDGES * sizeof(long long),
                   hipMemcpyDeviceToDevice, stream);
}
